// NG_GCNConv_15238543966833
// MI455X (gfx1250) — matrix operand order repaired and checked by reading, not compiled, not run on hardware
//
#include <hip/hip_runtime.h>
#include <hip/hip_bf16.h>
#include <math.h>

// ---------------------------------------------------------------------------
// Constants from the reference
// ---------------------------------------------------------------------------
#define FDIM   128
#define NITER  3
#define NGRAPH 256

typedef _Float16 v16h  __attribute__((ext_vector_type(16)));
typedef _Float16 half8 __attribute__((ext_vector_type(8)));
typedef float    v8f   __attribute__((ext_vector_type(8)));

// order-preserving float <-> uint map for atomic segment-max
__device__ __forceinline__ unsigned f2ord(float f) {
    unsigned u = __float_as_uint(f);
    return (u & 0x80000000u) ? ~u : (u | 0x80000000u);
}
__device__ __forceinline__ float ord2f(unsigned u) {
    return (u & 0x80000000u) ? __uint_as_float(u & 0x7FFFFFFFu)
                             : __uint_as_float(~u);
}

// ---------------------------------------------------------------------------
// Elementwise / setup kernels
// ---------------------------------------------------------------------------
__global__ void cvt_f32_to_f16(const float* __restrict__ src,
                               _Float16* __restrict__ dst, int n) {
    int i = blockIdx.x * blockDim.x + threadIdx.x;
    if (i < n) dst[i] = (_Float16)src[i];
}

// Pack W_gcn[iters][128][128] (row-major KxN) into the exact WMMA B-fragment
// layout so the GEMM can read each fragment as one contiguous 32B chunk:
//   Wpk[((it*8 + nt)*4 + kc)*32 + lane][j]  =  (f16) W[it][kc*32 + (j<8 ? hl*8 + j : 16 + hl*8 + j-8)][nt*16 + l16]
// where hl=lane>>4, l16=lane&15.  (ISA 7.12.2: B 32x16 f16, lane=N, K striped)
__global__ void pack_B(const float* __restrict__ W,
                       _Float16* __restrict__ Wpk, int total) {
    int i = blockIdx.x * blockDim.x + threadIdx.x;
    if (i >= total) return;
    int it   = i >> 14;            // / (128*128)
    int r    = i & 16383;
    int nt   = r >> 11;            // 0..7
    int kc   = (r >> 9) & 3;       // 0..3
    int lane = (r >> 4) & 31;
    int j    = r & 15;
    int hl   = lane >> 4;
    int col  = nt * 16 + (lane & 15);
    int k    = kc * 32 + (j < 8 ? hl * 8 + j : 16 + hl * 8 + (j - 8));
    Wpk[i] = (_Float16)W[(size_t)it * FDIM * FDIM + (size_t)k * FDIM + col];
}

__global__ void init_const(float* __restrict__ p, float v, int n) {
    int i = blockIdx.x * blockDim.x + threadIdx.x;
    if (i < n) p[i] = v;
}

__global__ void count_deg(const int* __restrict__ dst_idx,
                          float* __restrict__ deg, int E) {
    int e = blockIdx.x * blockDim.x + threadIdx.x;
    if (e < E) atomicAdd(&deg[dst_idx[e]], 1.0f);
}

__global__ void deg_to_dinv(float* __restrict__ d, int n) {
    int i = blockIdx.x * blockDim.x + threadIdx.x;
    if (i < n) {
        float v = d[i];
        d[i] = (v > 0.0f) ? rsqrtf(v) : 0.0f;
    }
}

// gcn[n*F + f] = bias[f]
__global__ void init_bias(float* __restrict__ gcn,
                          const float* __restrict__ bias, int total) {
    int i = blockIdx.x * blockDim.x + threadIdx.x;
    if (i < total) gcn[i] = bias[i & (FDIM - 1)];
}

__global__ void relu_ip(float* __restrict__ p, int n) {
    int i = blockIdx.x * blockDim.x + threadIdx.x;
    if (i < n) p[i] = fmaxf(p[i], 0.0f);
}

// ---------------------------------------------------------------------------
// WMMA GEMM:  C[N,128] = A_f16[N,128] * B_f16[128,128]  (B pre-packed)
// One wave computes a 16x64 strip (4 N-tiles): A fragment loaded once per
// K-step and reused across 4 WMMAs.  VGPR layouts per CDNA5 ISA 7.12.2.
// ---------------------------------------------------------------------------
__global__ void gemm_wmma_f16(const _Float16* __restrict__ A,
                              const _Float16* __restrict__ Bpk,
                              float* __restrict__ C,
                              int Nnodes, int mtiles) {
    const int wave = threadIdx.x >> 5;
    const int lane = threadIdx.x & 31;
    const int gid  = blockIdx.x * (blockDim.x >> 5) + wave;
    const int mt   = gid >> 1;        // 2 strips of 4 N-tiles per row
    const int ntg  = gid & 1;
    if (mt >= mtiles) return;

    const int hl  = lane >> 4;
    const int l16 = lane & 15;
    int row = mt * 16 + l16;
    if (row >= Nnodes) row = Nnodes - 1;   // duplicate-load guard (stores guarded)

    v8f acc[4] = {};
#pragma unroll
    for (int kc = 0; kc < 4; ++kc) {
        const int k0 = kc * 32;
        // A fragment: lanes 0-15 (M=l16) hold K={k0..k0+7, k0+16..k0+23},
        // lanes 16-31 hold K={k0+8..k0+15, k0+24..k0+31} -> two 16B loads.
        half8 lo = *(const half8*)(A + (size_t)row * FDIM + k0 + hl * 8);
        half8 hi = *(const half8*)(A + (size_t)row * FDIM + k0 + 16 + hl * 8);
        v16h a = __builtin_shufflevector(lo, hi, 0, 1, 2, 3, 4, 5, 6, 7,
                                         8, 9, 10, 11, 12, 13, 14, 15);
#pragma unroll
        for (int q = 0; q < 4; ++q) {
            const int nt = ntg * 4 + q;
            const v16h b = *(const v16h*)(Bpk + (((size_t)nt * 4 + kc) * 32 + lane) * 16);
            acc[q] = __builtin_amdgcn_wmma_f32_16x16x32_f16(
                /*neg_a=*/false, a, /*neg_b=*/false, b,
                /*c_mod=*/(short)0, acc[q], /*reuse_a=*/false, /*reuse_b=*/false);
        }
    }

#pragma unroll
    for (int q = 0; q < 4; ++q) {
        const int col = (ntg * 4 + q) * 16 + l16;
#pragma unroll
        for (int v = 0; v < 8; ++v) {
            int r = mt * 16 + v + hl * 8;
            if (r < Nnodes) C[(size_t)r * FDIM + col] = acc[q][v];
        }
    }
}

// ---------------------------------------------------------------------------
// GCN normalized scatter: for each edge (incl. self loops), one wave moves
// 128 floats (4 per lane) : gcn[dst] += h[src] * dinv[src]*dinv[dst]
// ---------------------------------------------------------------------------
__global__ void scatter_gcn(const float* __restrict__ h,
                            const int* __restrict__ src,
                            const int* __restrict__ dst,
                            const float* __restrict__ dinv,
                            float* __restrict__ gcn,
                            int E, int Nnodes) {
    int t    = blockIdx.x * blockDim.x + threadIdx.x;
    int edge = t >> 5;
    int lane = t & 31;
    if (edge >= E + Nnodes) return;
    int s, d;
    if (edge < E) { s = src[edge]; d = dst[edge]; }
    else          { s = d = edge - E; }
    float nrm = dinv[s] * dinv[d];
    const float4 hv = *(const float4*)(h + (size_t)s * FDIM + lane * 4);
    float* out = gcn + (size_t)d * FDIM + lane * 4;
    atomicAdd(out + 0, hv.x * nrm);
    atomicAdd(out + 1, hv.y * nrm);
    atomicAdd(out + 2, hv.z * nrm);
    atomicAdd(out + 3, hv.w * nrm);
}

// plain neighbor-sum scatter (no self loops, no norm): agg[dst] += gcn[src]
__global__ void scatter_agg(const float* __restrict__ gcn,
                            const int* __restrict__ src,
                            const int* __restrict__ dst,
                            float* __restrict__ agg, int E) {
    int t    = blockIdx.x * blockDim.x + threadIdx.x;
    int edge = t >> 5;
    int lane = t & 31;
    if (edge >= E) return;
    int s = src[edge], d = dst[edge];
    const float4 hv = *(const float4*)(gcn + (size_t)s * FDIM + lane * 4);
    float* out = agg + (size_t)d * FDIM + lane * 4;
    atomicAdd(out + 0, hv.x);
    atomicAdd(out + 1, hv.y);
    atomicAdd(out + 2, hv.z);
    atomicAdd(out + 3, hv.w);
}

// xc[n] = agg[n]·W_rel + gcn[n]·W_root + b_att   (one wave per node)
__global__ void attn_score(const float* __restrict__ agg,
                           const float* __restrict__ gcn,
                           const float* __restrict__ Wrel,
                           const float* __restrict__ Wroot,
                           const float* __restrict__ batt,
                           float* __restrict__ xc, int Nnodes) {
    int t    = blockIdx.x * blockDim.x + threadIdx.x;
    int node = t >> 5;
    int lane = t & 31;
    if (node >= Nnodes) return;
    float p = 0.0f;
    const float* ap = agg + (size_t)node * FDIM + lane * 4;
    const float* gp = gcn + (size_t)node * FDIM + lane * 4;
#pragma unroll
    for (int j = 0; j < 4; ++j)
        p += ap[j] * Wrel[lane * 4 + j] + gp[j] * Wroot[lane * 4 + j];
#pragma unroll
    for (int off = 16; off > 0; off >>= 1)
        p += __shfl_xor(p, off, 32);
    if (lane == 0) xc[node] = p + batt[0];
}

__global__ void seg_max(const float* __restrict__ xc,
                        const int* __restrict__ batch,
                        unsigned* __restrict__ mord, int Nnodes) {
    int n = blockIdx.x * blockDim.x + threadIdx.x;
    if (n < Nnodes) atomicMax(&mord[batch[n]], f2ord(xc[n]));
}

// xc[n] = exp(xc[n]-m[b]) ; den[b] += xc[n]
__global__ void seg_exp_den(float* __restrict__ xc,
                            const int* __restrict__ batch,
                            const unsigned* __restrict__ mord,
                            float* __restrict__ den, int Nnodes) {
    int n = blockIdx.x * blockDim.x + threadIdx.x;
    if (n < Nnodes) {
        int b = batch[n];
        float e = __expf(xc[n] - ord2f(mord[b]));
        xc[n] = e;
        atomicAdd(&den[b], e);
    }
}

// gx[batch[n], f] += gcn[n,f] * (xc[n]/den[batch[n]])   (thread per (n,f))
__global__ void pool_gx(const float* __restrict__ gcn,
                        const float* __restrict__ xc,
                        const float* __restrict__ den,
                        const int* __restrict__ batch,
                        float* __restrict__ gx, int total) {
    int i = blockIdx.x * blockDim.x + threadIdx.x;
    if (i >= total) return;
    int n = i >> 7;            // /128
    int f = i & (FDIM - 1);
    int b = batch[n];
    float sc = xc[n] / den[b];
    atomicAdd(&gx[(size_t)b * FDIM + f], gcn[(size_t)n * FDIM + f] * sc);
}

// gout[g,f] = tanh(gx[g]·W_gout[:,f] + b_gout[f])   (thread per (g,f))
__global__ void graph_out(const float* __restrict__ gx,
                          const float* __restrict__ Wg,
                          const float* __restrict__ bg,
                          float* __restrict__ gout) {
    int i = blockIdx.x * blockDim.x + threadIdx.x;
    if (i >= NGRAPH * FDIM) return;
    int g = i >> 7;
    int f = i & (FDIM - 1);
    const float* gp = gx + (size_t)g * FDIM;
    float acc = bg[f];
#pragma unroll 4
    for (int k = 0; k < FDIM; ++k)
        acc += gp[k] * Wg[(size_t)k * FDIM + f];
    gout[i] = tanhf(acc);
}

// scores[g,t] = softmax_t( sum_f gout_all[t,g,f]*a[f,t] + a_bias[t] )
__global__ void iter_scores(const float* __restrict__ gout_all,
                            const float* __restrict__ a,
                            const float* __restrict__ abias,
                            float* __restrict__ scores) {
    int g = blockIdx.x * blockDim.x + threadIdx.x;
    if (g >= NGRAPH) return;
    float raw[NITER];
#pragma unroll
    for (int t = 0; t < NITER; ++t) {
        float acc = abias[t];
        const float* gp = gout_all + (size_t)t * NGRAPH * FDIM + (size_t)g * FDIM;
        for (int f = 0; f < FDIM; ++f) acc += gp[f] * a[f * NITER + t];
        raw[t] = acc;
    }
    float m = fmaxf(raw[0], fmaxf(raw[1], raw[2]));
    float s = 0.0f;
#pragma unroll
    for (int t = 0; t < NITER; ++t) { raw[t] = __expf(raw[t] - m); s += raw[t]; }
    float inv = 1.0f / s;
#pragma unroll
    for (int t = 0; t < NITER; ++t) scores[g * NITER + t] = raw[t] * inv;
}

// out[n,f] = sum_t out_all[t,n,f] * scores[batch[n],t]
__global__ void combine_out(const float* __restrict__ out_all,
                            const float* __restrict__ scores,
                            const int* __restrict__ batch,
                            float* __restrict__ out,
                            int Nnodes, int total) {
    int i = blockIdx.x * blockDim.x + threadIdx.x;
    if (i >= total) return;
    int n = i >> 7;
    int b = batch[n];
    size_t NF = (size_t)Nnodes * FDIM;
    float acc = 0.0f;
#pragma unroll
    for (int t = 0; t < NITER; ++t)
        acc += out_all[(size_t)t * NF + i] * scores[b * NITER + t];
    out[i] = acc;
}

// ---------------------------------------------------------------------------
// Host-side launch
// ---------------------------------------------------------------------------
static inline size_t alignUp(size_t v, size_t a) { return (v + a - 1) & ~(a - 1); }

extern "C" void kernel_launch(void* const* d_in, const int* in_sizes, int n_in,
                              void* d_out, int out_size, void* d_ws, size_t ws_size,
                              hipStream_t stream) {
    const float* x      = (const float*)d_in[0];
    const int*   eidx   = (const int*)d_in[1];
    const int*   batch  = (const int*)d_in[2];
    const float* W_gcn  = (const float*)d_in[3];
    const float* b_gcn  = (const float*)d_in[4];
    const float* W_rel  = (const float*)d_in[5];
    const float* W_root = (const float*)d_in[6];
    const float* b_att  = (const float*)d_in[7];
    const float* W_gout = (const float*)d_in[8];
    const float* b_gout = (const float*)d_in[9];
    const float* a      = (const float*)d_in[10];
    const float* a_bias = (const float*)d_in[11];

    const int E = in_sizes[1] / 2;
    const int N = in_sizes[2];
    const int* src = eidx;
    const int* dst = eidx + E;
    const size_t NF = (size_t)N * FDIM;

    // ---- workspace carving -------------------------------------------------
    char* ws = (char*)d_ws;
    size_t off = 0;
    _Float16* xh = (_Float16*)(ws + off); off = alignUp(off + NF * 2, 256);
    _Float16* wpk = (_Float16*)(ws + off); off = alignUp(off + (size_t)NITER * FDIM * FDIM * 2, 256);
    float* dinv  = (float*)(ws + off);    off = alignUp(off + (size_t)N * 4, 256);
    float* h     = (float*)(ws + off);    off = alignUp(off + NF * 4, 256);   // GEMM out, reused as agg
    float* outA  = (float*)(ws + off);    off = alignUp(off + (size_t)NITER * NF * 4, 256);
    float* xc    = (float*)(ws + off);    off = alignUp(off + (size_t)N * 4, 256);
    unsigned* mo = (unsigned*)(ws + off); off = alignUp(off + (size_t)NGRAPH * 4, 256);
    float* den   = (float*)(ws + off);    off = alignUp(off + (size_t)NGRAPH * 4, 256);
    float* gx    = (float*)(ws + off);    off = alignUp(off + (size_t)NGRAPH * FDIM * 4, 256);
    float* goutA = (float*)(ws + off);    off = alignUp(off + (size_t)NITER * NGRAPH * FDIM * 4, 256);
    float* scor  = (float*)(ws + off);    off = alignUp(off + (size_t)NGRAPH * NITER * 4, 256);
    (void)ws_size;

    const int TB = 256;
    auto blk = [](long long n, int tb) { return (unsigned)((n + tb - 1) / tb); };

    // ---- setup: f16 copies (B in WMMA fragment layout), sym-norm degree ----
    cvt_f32_to_f16<<<blk(NF, TB), TB, 0, stream>>>(x, xh, (int)NF);
    pack_B<<<blk((long long)NITER * FDIM * FDIM, TB), TB, 0, stream>>>(
        W_gcn, wpk, NITER * FDIM * FDIM);
    init_const<<<blk(N, TB), TB, 0, stream>>>(dinv, 1.0f, N);   // self-loop
    count_deg<<<blk(E, TB), TB, 0, stream>>>(dst, dinv, E);
    deg_to_dinv<<<blk(N, TB), TB, 0, stream>>>(dinv, N);

    const int mtiles = (N + 15) / 16;
    const long long gemm_waves = (long long)mtiles * 2;   // 16x64 strip per wave

    for (int it = 0; it < NITER; ++it) {
        float* gcn = outA + (size_t)it * NF;

        // h = x @ W_gcn[it]  (WMMA)
        gemm_wmma_f16<<<blk(gemm_waves, 4), 128, 0, stream>>>(
            xh, wpk + (size_t)it * FDIM * FDIM, h, N, mtiles);

        // gcn = relu(scatter(norm*h) + bias)
        init_bias<<<blk(NF, TB), TB, 0, stream>>>(gcn, b_gcn + it * FDIM, (int)NF);
        scatter_gcn<<<blk((long long)(E + N) * 32, TB), TB, 0, stream>>>(
            h, src, dst, dinv, gcn, E, N);
        relu_ip<<<blk(NF, TB), TB, 0, stream>>>(gcn, (int)NF);

        // agg = neighbor sum (reuse h)
        hipMemsetAsync(h, 0, NF * 4, stream);
        scatter_agg<<<blk((long long)E * 32, TB), TB, 0, stream>>>(gcn, src, dst, h, E);

        // attention scores + segment softmax
        attn_score<<<blk((long long)N * 32, TB), TB, 0, stream>>>(
            h, gcn, W_rel, W_root, b_att, xc, N);
        hipMemsetAsync(mo, 0, NGRAPH * 4, stream);
        hipMemsetAsync(den, 0, NGRAPH * 4, stream);
        seg_max<<<blk(N, TB), TB, 0, stream>>>(xc, batch, mo, N);
        seg_exp_den<<<blk(N, TB), TB, 0, stream>>>(xc, batch, mo, den, N);

        // graph pooling + output transform
        hipMemsetAsync(gx, 0, (size_t)NGRAPH * FDIM * 4, stream);
        pool_gx<<<blk(NF, TB), TB, 0, stream>>>(gcn, xc, den, batch, gx, (int)NF);
        graph_out<<<blk(NGRAPH * FDIM, TB), TB, 0, stream>>>(
            gx, W_gout, b_gout, goutA + (size_t)it * NGRAPH * FDIM);
    }

    // iteration attention softmax + final combine
    iter_scores<<<blk(NGRAPH, TB), TB, 0, stream>>>(goutA, a, a_bias, scor);
    combine_out<<<blk(NF, TB), TB, 0, stream>>>(outA, scor, batch, (float*)d_out,
                                                N, (int)NF);
}
